// EdgeEncoder_28071906247258
// MI455X (gfx1250) — matrix operand order repaired and checked by reading, not compiled, not run on hardware
//
#include <hip/hip_runtime.h>
#include <stdint.h>

typedef __attribute__((ext_vector_type(16))) _Float16 v16h;
typedef __attribute__((ext_vector_type(8)))  float    v8f;

#define TM   32    // edges per tile = 2 WMMA M-tiles
#define XSTR 264   // halfs per x row: 256 data + 8 pad  (528 B -> bank stride 4)
#define HSTR 136   // halfs per h row: 128 data + 8 pad  (272 B -> bank stride 4)
#define OSTR 132   // floats per out row: 128 data + 4 pad

union FragH { v16h v; _Float16 e[16]; uint4 q[2]; };
union FragF { v8f  v; float    e[8]; };
union H4    { _Float16 e[4]; uint2 u2; };

__global__ __launch_bounds__(256) void edge_mlp_wmma_kernel(
    const float* __restrict__ node_emb,
    const int*   __restrict__ edge_index,
    const float* __restrict__ edge_sel,
    const float* __restrict__ W1,
    const float* __restrict__ b1,
    const float* __restrict__ W2,
    const float* __restrict__ b2,
    float* __restrict__ out,
    int N, int E, int BATCH)
{
    __shared__ _Float16 xs[TM * XSTR];   // concat(u,v) rows, f16
    __shared__ _Float16 hs[TM * HSTR];   // hidden rows, f16
    __shared__ float    outs[TM * OSTR]; // output staging, f32
    __shared__ float    sels[TM];

    const int lane = threadIdx.x & 31;
    const int wave = threadIdx.x >> 5;        // 0..7 -> owns output N-tile
    const int hsel = (lane >> 4) & 1;         // half-wave (K split)
    const int col  = wave * 16 + (lane & 15); // output column 0..127

    // ---- hoist W1/W2 B-fragments into registers (f32 -> f16) ----
    // B k order chosen here (contiguous, matches A loads): VGPR v, lanes 0-15 hold K=2v,2v+1 for N=lane;
    // lanes 16-31 hold K=16+2v,17+2v.
    FragH B1f[8];
    FragH B2f[4];
#pragma unroll
    for (int c = 0; c < 8; ++c)
#pragma unroll
        for (int v = 0; v < 8; ++v) {
            int k0 = c * 32 + hsel * 16 + 2 * v;
            B1f[c].e[2 * v]     = (_Float16)W1[(size_t)k0 * 128 + col];
            B1f[c].e[2 * v + 1] = (_Float16)W1[(size_t)(k0 + 1) * 128 + col];
        }
#pragma unroll
    for (int c = 0; c < 4; ++c)
#pragma unroll
        for (int v = 0; v < 8; ++v) {
            int k0 = c * 32 + hsel * 16 + 2 * v;
            B2f[c].e[2 * v]     = (_Float16)W2[(size_t)k0 * 128 + col];
            B2f[c].e[2 * v + 1] = (_Float16)W2[(size_t)(k0 + 1) * 128 + col];
        }
    const float b1c = b1[col];
    const float w1x = W1[(size_t)256 * 128 + col];  // EXTRA column of W1
    const float b2c = b2[col];

    const int tilesPerBatch = E / TM;
    const int totalTiles    = BATCH * tilesPerBatch;

    for (int tile = blockIdx.x; tile < totalTiles; tile += gridDim.x) {
        const int b  = tile / tilesPerBatch;
        const int e0 = (tile - b * tilesPerBatch) * TM;

        // ---- gather u/v node rows -> LDS (f32 -> f16), 128B/thread coalesced ----
        {
            const int t     = threadIdx.x;
            const int row   = t >> 2;      // 0..63 (32 u-rows + 32 v-rows)
            const int m     = row & 31;    // edge within tile
            const int which = row >> 5;    // 0 = u, 1 = v
            int idx = edge_index[((size_t)(b * E + e0 + m) << 1) + which];
            idx = idx < 0 ? 0 : (idx >= N ? N - 1 : idx);
            const float4* src =
                (const float4*)(node_emb + ((size_t)b * N + idx) * 128) + (t & 3) * 8;
#pragma unroll
            for (int i = 0; i < 8; ++i) {
                float4 f = src[i];
                H4 h4;
                h4.e[0] = (_Float16)f.x; h4.e[1] = (_Float16)f.y;
                h4.e[2] = (_Float16)f.z; h4.e[3] = (_Float16)f.w;
                *(uint2*)&xs[m * XSTR + which * 128 + (t & 3) * 32 + i * 4] = h4.u2;
            }
            if (which == 0 && (t & 3) == 0)
                sels[m] = edge_sel[(size_t)b * E + e0 + m];
        }
        __syncthreads();

        // ---- GEMM1: x[32x256] @ W1[256 x 16-col tile]; two independent M-tiles ----
        FragF acc[2];
#pragma unroll
        for (int mt = 0; mt < 2; ++mt)
#pragma unroll
            for (int r = 0; r < 8; ++r) {
                int m = mt * 16 + r + hsel * 8;   // C layout: M = r (+8 for hi lanes)
                acc[mt].e[r] = b1c + sels[m] * w1x;
            }
#pragma unroll
        for (int c = 0; c < 8; ++c) {
            FragH a0, a1;  // A layout: lane M=lane&15; lo lanes K+0..15, hi K+16..31
            const char* p0 = (const char*)&xs[(lane & 15) * XSTR + c * 32 + hsel * 16];
            const char* p1 = p0 + 16 * XSTR * 2;  // second M-tile, rows +16
            a0.q[0] = *(const uint4*)p0;
            a0.q[1] = *(const uint4*)(p0 + 16);
            a1.q[0] = *(const uint4*)p1;
            a1.q[1] = *(const uint4*)(p1 + 16);
            acc[0].v = __builtin_amdgcn_wmma_f32_16x16x32_f16(
                false, a0.v, false, B1f[c].v, (short)0, acc[0].v, false, false);
            acc[1].v = __builtin_amdgcn_wmma_f32_16x16x32_f16(
                false, a1.v, false, B1f[c].v, (short)0, acc[1].v, false, false);
        }
#pragma unroll
        for (int mt = 0; mt < 2; ++mt)
#pragma unroll
            for (int r = 0; r < 8; ++r) {
                int m = mt * 16 + r + hsel * 8;
                hs[m * HSTR + col] = (_Float16)fmaxf(acc[mt].e[r], 0.0f);
            }
        __syncthreads();

        // ---- GEMM2: h[32x128] @ W2[128 x 16-col tile] ----
        FragF acc2[2];
#pragma unroll
        for (int mt = 0; mt < 2; ++mt)
#pragma unroll
            for (int r = 0; r < 8; ++r) acc2[mt].e[r] = b2c;
#pragma unroll
        for (int c = 0; c < 4; ++c) {
            FragH a0, a1;
            const char* p0 = (const char*)&hs[(lane & 15) * HSTR + c * 32 + hsel * 16];
            const char* p1 = p0 + 16 * HSTR * 2;
            a0.q[0] = *(const uint4*)p0;
            a0.q[1] = *(const uint4*)(p0 + 16);
            a1.q[0] = *(const uint4*)p1;
            a1.q[1] = *(const uint4*)(p1 + 16);
            acc2[0].v = __builtin_amdgcn_wmma_f32_16x16x32_f16(
                false, a0.v, false, B2f[c].v, (short)0, acc2[0].v, false, false);
            acc2[1].v = __builtin_amdgcn_wmma_f32_16x16x32_f16(
                false, a1.v, false, B2f[c].v, (short)0, acc2[1].v, false, false);
        }
#pragma unroll
        for (int mt = 0; mt < 2; ++mt)
#pragma unroll
            for (int r = 0; r < 8; ++r) {
                int m = mt * 16 + r + hsel * 8;
                outs[m * OSTR + col] = fmaxf(acc2[mt].e[r], 0.0f);
            }
        __syncthreads();

        // ---- LDS -> global via async DMA path, fully coalesced b128 ----
        {
            const int t   = threadIdx.x;
            const int row = t >> 3;   // 0..31
            const int f4  = t & 7;    // float4 slot within row
#pragma unroll
            for (int i = 0; i < 4; ++i) {
                uint64_t ga = (uint64_t)(uintptr_t)(
                    out + (size_t)(b * E + e0 + row) * 128 + (f4 + i * 8) * 4);
                // low 32 bits of flat shared address == LDS byte offset
                uint32_t la = (uint32_t)(uintptr_t)&outs[row * OSTR + (f4 + i * 8) * 4];
                asm volatile("global_store_async_from_lds_b128 %0, %1, off"
                             :: "v"(ga), "v"(la) : "memory");
            }
            asm volatile("s_wait_asynccnt 0x0" ::: "memory");
        }
        __syncthreads();
    }
}

extern "C" void kernel_launch(void* const* d_in, const int* in_sizes, int n_in,
                              void* d_out, int out_size, void* d_ws, size_t ws_size,
                              hipStream_t stream) {
    const float* node_emb   = (const float*)d_in[0];
    const int*   edge_index = (const int*)d_in[1];
    const float* edge_sel   = (const float*)d_in[2];
    const float* W1         = (const float*)d_in[3];
    const float* b1         = (const float*)d_in[4];
    const float* W2         = (const float*)d_in[5];
    const float* b2         = (const float*)d_in[6];
    float*       out        = (float*)d_out;

    const int BATCH = 4;
    const int E = in_sizes[2] / BATCH;            // edge_sel is [B,E,1]
    const int N = in_sizes[0] / (BATCH * 128);    // node_emb is [B,N,128]

    const int totalTiles = BATCH * (E / TM);
    int grid = totalTiles < 2048 ? totalTiles : 2048;

    edge_mlp_wmma_kernel<<<grid, 256, 0, stream>>>(
        node_emb, edge_index, edge_sel, W1, b1, W2, b2, out, N, E, BATCH);
}
